// MoE_3530463117618
// MI455X (gfx1250) — matrix operand order repaired and checked by reading, compiled, not run on hardware
//
#include <hip/hip_runtime.h>

// ---------------- problem constants ----------------
constexpr int Bq = 4, Sq = 2048, D = 2048, E = 8, FF = 512, TOPK = 2;
constexpr int T   = Bq * Sq;          // 8192 tokens
constexpr int NA  = T * TOPK;         // 16384 routed assignments
constexpr int NB  = NA / 256;         // 64 blocks for hist/assign
constexpr int PADROWS = NA + E * 128; // 17408 worst-case padded rows
constexpr int MAX_RT  = PADROWS / 128;// 136 row tiles of 128

// ---------------- WMMA vector types ----------------
typedef __attribute__((ext_vector_type(16))) __bf16 v16bf;
typedef __attribute__((ext_vector_type(8)))  float  v8f;

union Frag {
    v16bf v;
    uint4 q[2];
    unsigned short s[16];
};

__device__ __forceinline__ unsigned short f2bf(float f) {
    union { float f; unsigned u; } v; v.f = f;
    unsigned r = v.u + 0x7FFFu + ((v.u >> 16) & 1u); // round-to-nearest-even
    return (unsigned short)(r >> 16);
}

__device__ __forceinline__ v8f zero8() {
    v8f z = {0.f, 0.f, 0.f, 0.f, 0.f, 0.f, 0.f, 0.f};
    return z;
}

// A fragment (16x32 bf16) from an fp32 row: lane<16 -> K {0..7,16..23} of row m,
// lane>=16 -> K {8..15,24..31}. Two 32B fp32 chunks, converted to bf16.
__device__ __forceinline__ void loadA_f32(Frag& a, const float* __restrict__ arow,
                                          int k0, int kh) {
    const float* p1 = arow + k0 + 8 * kh;
    const float* p2 = arow + k0 + 16 + 8 * kh;
    #pragma unroll
    for (int q = 0; q < 8; ++q) { a.s[q] = f2bf(p1[q]); a.s[8 + q] = f2bf(p2[q]); }
}

// A fragment from a bf16 row (two contiguous 16B chunks)
__device__ __forceinline__ void loadA_bf16(Frag& a, const unsigned short* __restrict__ arow,
                                           int k0, int kh) {
    a.q[0] = *(const uint4*)(arow + k0 + 8 * kh);
    a.q[1] = *(const uint4*)(arow + k0 + 16 + 8 * kh);
}

// ---------------- utility kernels ----------------
__global__ __launch_bounds__(256) void zero_out_kernel(float4* __restrict__ p, long n4) {
    long i = (long)blockIdx.x * blockDim.x + threadIdx.x;
    if (i < n4) p[i] = make_float4(0.f, 0.f, 0.f, 0.f);
}

__global__ __launch_bounds__(256) void init_rows_kernel(int* __restrict__ rtok,
                                                        float* __restrict__ rscore) {
    int i = blockIdx.x * 256 + threadIdx.x;
    if (i < PADROWS) { rtok[i] = -1; rscore[i] = 0.f; }
}

// fp32 (Kdim x Ndim per expert, row-major) -> bf16 pairs in exact WMMA
// B-fragment order: idx = (((kt*ntiles + nt)*32 + lane)*8 + j)
__global__ __launch_bounds__(256) void swizzle_kernel(const float* __restrict__ w,
                                                      unsigned* __restrict__ outw,
                                                      int Kdim, int Ndim) {
    long idx = (long)blockIdx.x * 256 + threadIdx.x;
    long per = (long)Kdim * Ndim / 2;
    if (idx >= (long)E * per) return;
    int  e = (int)(idx / per);
    long u = idx % per;
    int  j    = (int)(u & 7);
    int  lane = (int)((u >> 3) & 31);
    long rest = u >> 8;
    int  ntiles = Ndim / 16;
    int  nt = (int)(rest % ntiles);
    int  kt = (int)(rest / ntiles);
    int  k = kt * 32 + (j >> 2) * 16 + (lane >> 4) * 8 + 2 * (j & 3);
    int  n = nt * 16 + (lane & 15);
    const float* we = w + (size_t)e * Kdim * Ndim;
    unsigned lo = f2bf(we[(size_t)k * Ndim + n]);
    unsigned hi = f2bf(we[(size_t)(k + 1) * Ndim + n]);
    outw[idx] = lo | (hi << 16);
}

// gather one padded row of x into bf16 (zeros for padding rows): one block per row
__global__ __launch_bounds__(256) void gather_rows_kernel(const float* __restrict__ x,
                                                          const int* __restrict__ rtok,
                                                          const int* __restrict__ poff,
                                                          unsigned short* __restrict__ xg) {
    int row = blockIdx.x;
    if (row >= poff[E]) return;
    int tok = rtok[row];
    unsigned short* dst = xg + (size_t)row * D;
    if (tok < 0) {
        // padding row: write zeros (h becomes 0; row is never scattered anyway)
        for (int d = threadIdx.x * 8; d < D; d += 256 * 8) {
            uint4 z = make_uint4(0u, 0u, 0u, 0u);
            *(uint4*)(dst + d) = z;
        }
        return;
    }
    const float* src = x + (size_t)tok * D;
    for (int d = threadIdx.x * 8; d < D; d += 256 * 8) {
        float4 f0 = *(const float4*)(src + d);
        float4 f1 = *(const float4*)(src + d + 4);
        unsigned short s[8];
        s[0] = f2bf(f0.x); s[1] = f2bf(f0.y); s[2] = f2bf(f0.z); s[3] = f2bf(f0.w);
        s[4] = f2bf(f1.x); s[5] = f2bf(f1.y); s[6] = f2bf(f1.z); s[7] = f2bf(f1.w);
        *(uint4*)(dst + d) = *(const uint4*)s;
    }
}

// ---------------- router: one wave32 per token ----------------
__global__ __launch_bounds__(256) void router_kernel(const float* __restrict__ x,
                                                     const float* __restrict__ rw,
                                                     int* __restrict__ topki,
                                                     float* __restrict__ topkp) {
    int wave = (blockIdx.x * blockDim.x + threadIdx.x) >> 5;
    int lane = threadIdx.x & 31;
    if (wave >= T) return;
    const float* xr = x + (size_t)wave * D;
    float acc[E];
    #pragma unroll
    for (int e = 0; e < E; ++e) acc[e] = 0.f;
    for (int d = lane; d < D; d += 32) {
        float xv = xr[d];
        #pragma unroll
        for (int e = 0; e < E; ++e) acc[e] += xv * rw[e * D + d];
    }
    #pragma unroll
    for (int e = 0; e < E; ++e) {
        #pragma unroll
        for (int m = 16; m >= 1; m >>= 1) acc[e] += __shfl_xor(acc[e], m, 32);
    }
    if (lane == 0) {
        int i0 = 0; float s0 = acc[0];
        #pragma unroll
        for (int e = 1; e < E; ++e) if (acc[e] > s0) { s0 = acc[e]; i0 = e; }
        int i1 = -1; float s1 = -3.4e38f;
        #pragma unroll
        for (int e = 0; e < E; ++e) if (e != i0 && acc[e] > s1) { s1 = acc[e]; i1 = e; }
        float p1 = 1.f / (1.f + __expf(s0 - s1)); // softmax over [s0, s1], s0 >= s1
        float p0 = 1.f - p1;
        topki[wave * 2 + 0] = i0; topki[wave * 2 + 1] = i1;
        topkp[wave * 2 + 0] = p0; topkp[wave * 2 + 1] = p1;
    }
}

// ---------------- deterministic expert sort ----------------
__global__ __launch_bounds__(256) void hist_kernel(const int* __restrict__ topki,
                                                   int* __restrict__ bhist) {
    __shared__ int h[E];
    if (threadIdx.x < E) h[threadIdx.x] = 0;
    __syncthreads();
    int i = blockIdx.x * 256 + threadIdx.x;
    atomicAdd(&h[topki[i]], 1);
    __syncthreads();
    if (threadIdx.x < E) bhist[blockIdx.x * E + threadIdx.x] = h[threadIdx.x];
}

__global__ void scan_kernel(const int* __restrict__ bhist, int* __restrict__ poff,
                            int* __restrict__ bbase) {
    if (threadIdx.x != 0 || blockIdx.x != 0) return;
    int counts[E];
    for (int e = 0; e < E; ++e) counts[e] = 0;
    for (int b = 0; b < NB; ++b)
        for (int e = 0; e < E; ++e) counts[e] += bhist[b * E + e];
    int off = 0;
    for (int e = 0; e < E; ++e) {
        poff[e] = off;
        off += (counts[e] + 127) & ~127;   // pad each expert segment to 128 rows
    }
    poff[E] = off;
    int run[E];
    for (int e = 0; e < E; ++e) run[e] = poff[e];
    for (int b = 0; b < NB; ++b)
        for (int e = 0; e < E; ++e) { bbase[b * E + e] = run[e]; run[e] += bhist[b * E + e]; }
}

__global__ __launch_bounds__(256) void assign_kernel(const int* __restrict__ topki,
                                                     const float* __restrict__ topkp,
                                                     const int* __restrict__ bbase,
                                                     int* __restrict__ rtok,
                                                     float* __restrict__ rscore) {
    __shared__ int se[256];
    int i = blockIdx.x * 256 + threadIdx.x;
    int e = topki[i];
    se[threadIdx.x] = e;
    __syncthreads();
    int rank = 0;
    for (int j = 0; j < threadIdx.x; ++j) rank += (se[j] == e);
    int pos = bbase[blockIdx.x * E + e] + rank;
    rtok[pos]   = i >> 1;      // token id
    rscore[pos] = topkp[i];    // routing weight
}

// ---------------- GEMM 1&3 fused + SiLU: h = silu(x@w1) * (x@w3) ----------------
// grid (MAX_RT, FF/64), block 128 = 4 waves; wave w owns rows r0..r0+31 (two A
// fragments) x 64 cols (four B fragments shared by both A frags) -> 16 WMMA per
// k-step. USE_XG: A comes from pre-gathered bf16 rows (pure b128 loads);
// fallback: gather+convert fp32 x rows in-loop.
template <bool USE_XG>
__global__ __launch_bounds__(128) void gemm13_kernel(const float* __restrict__ x,
                                                     const unsigned short* __restrict__ xg,
                                                     const unsigned* __restrict__ w1s,
                                                     const unsigned* __restrict__ w3s,
                                                     const int* __restrict__ poff,
                                                     const int* __restrict__ rtok,
                                                     unsigned short* __restrict__ hbuf) {
    int wave = threadIdx.x >> 5, lane = threadIdx.x & 31;
    int rt = blockIdx.x;
    int total = poff[E];
    if (rt * 128 >= total) return;
    int r0 = rt * 128 + wave * 32;
    int e = 0;
    while (r0 >= poff[e + 1]) ++e;     // segments are 128-row multiples: no tile crossing
    int nb = blockIdx.y * 64;
    int ntb = nb >> 4;
    int m = lane & 15, kh = lane >> 4;

    const float* arow0_f = nullptr; const float* arow1_f = nullptr;
    const unsigned short* arow0_h = nullptr; const unsigned short* arow1_h = nullptr;
    if (USE_XG) {
        arow0_h = xg + (size_t)(r0 + m) * D;
        arow1_h = xg + (size_t)(r0 + 16 + m) * D;
    } else {
        int tok0 = rtok[r0 + m];
        int tok1 = rtok[r0 + 16 + m];
        arow0_f = x + (size_t)(tok0 < 0 ? 0 : tok0) * D;
        arow1_f = x + (size_t)(tok1 < 0 ? 0 : tok1) * D;
    }
    const unsigned* b1e = w1s + (size_t)e * (D * FF / 2);
    const unsigned* b3e = w3s + (size_t)e * (D * FF / 2);
    v8f acc1[2][4], acc3[2][4];
    #pragma unroll
    for (int mi = 0; mi < 2; ++mi)
        #pragma unroll
        for (int j = 0; j < 4; ++j) { acc1[mi][j] = zero8(); acc3[mi][j] = zero8(); }

    for (int kt = 0; kt < D / 32; ++kt) {
        int k0 = kt * 32;
        // ---- load everything for this k-step into distinct registers ----
        Frag b1f[4], b3f[4];
        #pragma unroll
        for (int j = 0; j < 4; ++j) {
            const unsigned* bp1 = b1e + ((size_t)(kt * (FF / 16) + ntb + j) * 32 + lane) * 8;
            b1f[j].q[0] = *(const uint4*)bp1; b1f[j].q[1] = *(const uint4*)(bp1 + 4);
            const unsigned* bp3 = b3e + ((size_t)(kt * (FF / 16) + ntb + j) * 32 + lane) * 8;
            b3f[j].q[0] = *(const uint4*)bp3; b3f[j].q[1] = *(const uint4*)(bp3 + 4);
        }
        Frag a0, a1;
        if (USE_XG) {
            loadA_bf16(a0, arow0_h, k0, kh);
            loadA_bf16(a1, arow1_h, k0, kh);
        } else {
            loadA_f32(a0, arow0_f, k0, kh);
            loadA_f32(a1, arow1_f, k0, kh);
        }
        // ---- WMMA burst ----
        #pragma unroll
        for (int j = 0; j < 4; ++j) {
            acc1[0][j] = __builtin_amdgcn_wmma_f32_16x16x32_bf16(false, a0.v, false, b1f[j].v,
                                                                 (short)0, acc1[0][j], false, false);
            acc1[1][j] = __builtin_amdgcn_wmma_f32_16x16x32_bf16(false, a1.v, false, b1f[j].v,
                                                                 (short)0, acc1[1][j], false, false);
            acc3[0][j] = __builtin_amdgcn_wmma_f32_16x16x32_bf16(false, a0.v, false, b3f[j].v,
                                                                 (short)0, acc3[0][j], false, false);
            acc3[1][j] = __builtin_amdgcn_wmma_f32_16x16x32_bf16(false, a1.v, false, b3f[j].v,
                                                                 (short)0, acc3[1][j], false, false);
        }
    }
    // epilogue: h = silu(x1) * x3  (C layout: VGPR v -> M = v + 8*(lane>>4), N = lane&15)
    #pragma unroll
    for (int mi = 0; mi < 2; ++mi) {
        #pragma unroll
        for (int j = 0; j < 4; ++j) {
            #pragma unroll
            for (int v = 0; v < 8; ++v) {
                float x1 = acc1[mi][j][v], x3 = acc3[mi][j][v];
                float hv = x1 * (1.f / (1.f + __expf(-x1))) * x3;
                int row = r0 + mi * 16 + v + 8 * kh;
                int n   = nb + j * 16 + m;
                hbuf[(size_t)row * FF + n] = f2bf(hv);
            }
        }
    }
}

// ---------------- GEMM 2 + weighted scatter: out[tok] += score * (h @ w2) ----------------
// grid (MAX_RT, D/64), block 128 = 4 waves; wave owns 32 rows x 64 cols.
__global__ __launch_bounds__(128) void gemm2_kernel(const unsigned short* __restrict__ hbuf,
                                                    const unsigned* __restrict__ w2s,
                                                    const int* __restrict__ poff,
                                                    const int* __restrict__ rtok,
                                                    const float* __restrict__ rscore,
                                                    float* __restrict__ out) {
    int wave = threadIdx.x >> 5, lane = threadIdx.x & 31;
    int rt = blockIdx.x;
    int total = poff[E];
    if (rt * 128 >= total) return;
    int r0 = rt * 128 + wave * 32;
    int e = 0;
    while (r0 >= poff[e + 1]) ++e;
    int nb = blockIdx.y * 64;
    int ntb = nb >> 4;
    int m = lane & 15, kh = lane >> 4;
    const unsigned short* arow0 = hbuf + (size_t)(r0 + m) * FF;
    const unsigned short* arow1 = hbuf + (size_t)(r0 + 16 + m) * FF;
    const unsigned* b2e = w2s + (size_t)e * (FF * D / 2);
    v8f acc[2][4];
    #pragma unroll
    for (int mi = 0; mi < 2; ++mi)
        #pragma unroll
        for (int j = 0; j < 4; ++j) acc[mi][j] = zero8();

    for (int kt = 0; kt < FF / 32; ++kt) {
        int k0 = kt * 32;
        Frag bf[4];
        #pragma unroll
        for (int j = 0; j < 4; ++j) {
            const unsigned* bp = b2e + ((size_t)(kt * (D / 16) + ntb + j) * 32 + lane) * 8;
            bf[j].q[0] = *(const uint4*)bp; bf[j].q[1] = *(const uint4*)(bp + 4);
        }
        Frag a0, a1;
        loadA_bf16(a0, arow0, k0, kh);
        loadA_bf16(a1, arow1, k0, kh);
        #pragma unroll
        for (int j = 0; j < 4; ++j) {
            acc[0][j] = __builtin_amdgcn_wmma_f32_16x16x32_bf16(false, a0.v, false, bf[j].v,
                                                                (short)0, acc[0][j], false, false);
            acc[1][j] = __builtin_amdgcn_wmma_f32_16x16x32_bf16(false, a1.v, false, bf[j].v,
                                                                (short)0, acc[1][j], false, false);
        }
    }
    #pragma unroll
    for (int mi = 0; mi < 2; ++mi) {
        #pragma unroll
        for (int j = 0; j < 4; ++j) {
            #pragma unroll
            for (int v = 0; v < 8; ++v) {
                int row = r0 + mi * 16 + v + 8 * kh;
                int tok = rtok[row];
                if (tok >= 0) {
                    float wsc = rscore[row];
                    atomicAdd(out + (size_t)tok * D + nb + j * 16 + m, wsc * acc[mi][j][v]);
                }
            }
        }
    }
}

// ---------------- host launch ----------------
extern "C" void kernel_launch(void* const* d_in, const int* in_sizes, int n_in,
                              void* d_out, int out_size, void* d_ws, size_t ws_size,
                              hipStream_t stream) {
    const float* x  = (const float*)d_in[0];
    const float* rw = (const float*)d_in[1];
    const float* w1 = (const float*)d_in[2];
    const float* w2 = (const float*)d_in[3];
    const float* w3 = (const float*)d_in[4];
    float* out = (float*)d_out;
    char* ws = (char*)d_ws;

    // workspace layout (bytes)
    constexpr size_t OFF_POFF   = 64;                       // int[9]
    constexpr size_t OFF_BHIST  = 128;                      // int[64*8]
    constexpr size_t OFF_BBASE  = 2304;                     // int[64*8]
    constexpr size_t OFF_TOPKI  = 4608;                     // int[NA]
    constexpr size_t OFF_TOPKP  = 70400;                    // float[NA]
    constexpr size_t OFF_RTOK   = 136192;                   // int[PADROWS]
    constexpr size_t OFF_RSCORE = 205824;                   // float[PADROWS]
    constexpr size_t OFF_H      = 275456;                   // bf16[PADROWS*FF]
    constexpr size_t OFF_W1S    = OFF_H   + (size_t)PADROWS * FF * 2;   // 18101248
    constexpr size_t OFF_W3S    = OFF_W1S + (size_t)E * D * FF * 2;     // 34878464
    constexpr size_t OFF_W2S    = OFF_W3S + (size_t)E * D * FF * 2;     // 51655680
    constexpr size_t OFF_XG     = OFF_W2S + (size_t)E * FF * D * 2;     // 68432896
    constexpr size_t NEED_XG    = OFF_XG  + (size_t)PADROWS * D * 2;    // ~139.7 MB

    int*      poff   = (int*)(ws + OFF_POFF);
    int*      bhist  = (int*)(ws + OFF_BHIST);
    int*      bbase  = (int*)(ws + OFF_BBASE);
    int*      topki  = (int*)(ws + OFF_TOPKI);
    float*    topkp  = (float*)(ws + OFF_TOPKP);
    int*      rtok   = (int*)(ws + OFF_RTOK);
    float*    rscore = (float*)(ws + OFF_RSCORE);
    unsigned short* hbuf = (unsigned short*)(ws + OFF_H);
    unsigned* w1s = (unsigned*)(ws + OFF_W1S);
    unsigned* w3s = (unsigned*)(ws + OFF_W3S);
    unsigned* w2s = (unsigned*)(ws + OFF_W2S);
    unsigned short* xg = (unsigned short*)(ws + OFF_XG);
    const bool use_xg = (ws_size >= NEED_XG);   // fixed per process -> deterministic

    // 1. zero output accumulator, init row maps
    long n4 = (long)T * D / 4;
    zero_out_kernel<<<dim3((unsigned)((n4 + 255) / 256)), 256, 0, stream>>>((float4*)out, n4);
    init_rows_kernel<<<dim3((PADROWS + 255) / 256), 256, 0, stream>>>(rtok, rscore);

    // 2. weight swizzle into WMMA B-fragment order (bf16 pairs)
    long swn = (long)E * D * FF / 2;  // same element count for all three tensors
    unsigned swg = (unsigned)((swn + 255) / 256);
    swizzle_kernel<<<dim3(swg), 256, 0, stream>>>(w1, w1s, D, FF);
    swizzle_kernel<<<dim3(swg), 256, 0, stream>>>(w3, w3s, D, FF);
    swizzle_kernel<<<dim3(swg), 256, 0, stream>>>(w2, w2s, FF, D);

    // 3. router (one wave per token) -> top-2 experts + softmax weights
    router_kernel<<<dim3(T / 8), 256, 0, stream>>>(x, rw, topki, topkp);

    // 4. deterministic expert-centric ordering with 128-row padded segments
    hist_kernel<<<dim3(NB), 256, 0, stream>>>(topki, bhist);
    scan_kernel<<<dim3(1), 32, 0, stream>>>(bhist, poff, bbase);
    assign_kernel<<<dim3(NB), 256, 0, stream>>>(topki, topkp, bbase, rtok, rscore);

    // 5. grouped SwiGLU up-projections (bf16 WMMA, fused SiLU*x3)
    if (use_xg) {
        // pre-gather tokens into bf16 padded-row order: A loads become pure b128
        gather_rows_kernel<<<dim3(PADROWS), 256, 0, stream>>>(x, rtok, poff, xg);
        gemm13_kernel<true><<<dim3(MAX_RT, FF / 64), 128, 0, stream>>>(x, xg, w1s, w3s,
                                                                       poff, rtok, hbuf);
    } else {
        gemm13_kernel<false><<<dim3(MAX_RT, FF / 64), 128, 0, stream>>>(x, xg, w1s, w3s,
                                                                        poff, rtok, hbuf);
    }

    // 6. grouped down-projection + weighted scatter-add into output
    gemm2_kernel<<<dim3(MAX_RT, D / 64), 128, 0, stream>>>(hbuf, w2s, poff, rtok, rscore, out);
}
